// MwAN_full_60687887892758
// MI455X (gfx1250) — compile-verified, matrix operand order checked by reading, not run on hardware
//
#include <hip/hip_runtime.h>
#include <hip/hip_fp16.h>

typedef __attribute__((ext_vector_type(16))) _Float16 v16h;
typedef __attribute__((ext_vector_type(8)))  float    v8f;

// ---------------------------------------------------------------------------
// WMMA helper: D = A(16x32 f16) * B(32x16 f16) + C(16x16 f32), wave32.
// ---------------------------------------------------------------------------
__device__ __forceinline__ v8f wmma_f16(v16h a, v16h b, v8f c) {
  return __builtin_amdgcn_wmma_f32_16x16x32_f16(
      /*neg_a=*/false, a, /*neg_b=*/false, b,
      /*c_mod=*/(short)0, c, /*reuse_a=*/false, /*reuse_b=*/false);
}

// ISA 7.12.2: 16-bit A 16x32 (MxK). Lanes 0-15 hold M=lane with K halves
// {0..7,16..23}; lanes 16-31 hold M=lane-16 with K halves {8..15,24..31}.
// B 32x16 assumed transpose-symmetric (N=lane&15, same K interleave).
__device__ __forceinline__ void frag_k_list(int lane, int* kk) {
  int base = (lane & 16) ? 8 : 0;
#pragma unroll
  for (int i = 0; i < 8; i++) { kk[i] = base + i; kk[8 + i] = 16 + base + i; }
}

__device__ __forceinline__ float actf(float x, int act) {
  switch (act) {
    case 1: return 1.f / (1.f + __expf(-x));   // sigmoid
    case 2: return tanhf(x);                   // tanh
    case 3: return x > 0.f ? x : 0.01f * x;    // leaky_relu
    default: return x;
  }
}

// CDNA5 async global->LDS copy of 64 contiguous bytes (4 x b128), tracked by
// ASYNCcnt (CDNA5 ISA). ldsoff = LDS byte offset (addr[31:0]
// of the shared-aperture address), ga = 64-bit global address. Both 16B aligned.
__device__ __forceinline__ void async_copy64(unsigned ldsoff, unsigned long long ga) {
  asm volatile(
      "global_load_async_to_lds_b128 %0, %1, off\n\t"
      "global_load_async_to_lds_b128 %0, %1, off offset:16\n\t"
      "global_load_async_to_lds_b128 %0, %1, off offset:32\n\t"
      "global_load_async_to_lds_b128 %0, %1, off offset:48"
      :: "v"(ldsoff), "v"(ga) : "memory");
}
__device__ __forceinline__ void async_wait0() {
  asm volatile("s_wait_asynccnt 0x0" ::: "memory");
}

// ---------------------------------------------------------------------------
// Generic batched GEMM: C = act(A @ B (+bias)); fp32 in/out, f16 WMMA core.
// One wave (32 threads) per 16x32 C block (2 accumulators share A fragment).
// Optional transB (B stored NxK). Fast interior paths use async-to-LDS for A.
// ---------------------------------------------------------------------------
__global__ void k_gemm(const float* __restrict__ A, const float* __restrict__ B,
                       const float* __restrict__ bias, float* __restrict__ C,
                       int M, int N, int K, int lda, int ldb, int ldc,
                       long sA, long sB, long sC, int transB, int act) {
  __shared__ __align__(16) float As[16][36];   // 144B row stride: 16B aligned
  __shared__ float Bs[32][33];
  int lane = threadIdx.x;
  int tn = blockIdx.x, tm = blockIdx.y, b = blockIdx.z;
  const float* Ab = A + (long)b * sA;
  const float* Bb = B + (long)b * sB;
  float* Cb = C + (long)b * sC;
  int row0 = tm * 16, col0 = tn * 32;
  int kk[16]; frag_k_list(lane, kk);
  int m_loc = lane & 15;
  v8f acc0 = {}, acc1 = {};
  bool mfull = (row0 + 16 <= M);
  bool nfull = (col0 + 32 <= N);
#pragma unroll 2
  for (int kc = 0; kc < K; kc += 32) {
    bool kfull = (kc + 32 <= K);
    // ---- stage A tile (16x32) ----
    if (mfull && kfull) {
      int r = lane >> 1, c0 = (lane & 1) * 16;
      const float* src = Ab + (long)(row0 + r) * lda + kc + c0;
      async_copy64((unsigned)(uintptr_t)&As[r][c0], (unsigned long long)(uintptr_t)src);
      __builtin_prefetch(src + 32, 0, 1);
    } else {
      for (int idx = lane; idx < 512; idx += 32) {
        int r = idx >> 5, c = idx & 31;
        int gr = row0 + r, gc = kc + c;
        As[r][c] = (gr < M && gc < K) ? Ab[(long)gr * lda + gc] : 0.f;
      }
    }
    // ---- stage B tile (32x32) ----
    if (kfull && nfull) {
      if (!transB) {
        const float* src = Bb + (long)(kc + lane) * ldb + col0;
#pragma unroll 8
        for (int c = 0; c < 32; c++) Bs[lane][c] = src[c];
        __builtin_prefetch(src + 32 * ldb, 0, 1);
      } else {
        const float* src = Bb + (long)(col0 + lane) * ldb + kc;
#pragma unroll 8
        for (int c = 0; c < 32; c++) Bs[c][lane] = src[c];
      }
    } else {
      for (int idx = lane; idx < 1024; idx += 32) {
        int r = idx >> 5, c = idx & 31;
        int gk = kc + r, gn = col0 + c;
        float v = 0.f;
        if (gk < K && gn < N)
          v = transB ? Bb[(long)gn * ldb + gk] : Bb[(long)gk * ldb + gn];
        Bs[r][c] = v;
      }
    }
    async_wait0();
    __syncthreads();
    v16h af, bf0, bf1;
#pragma unroll
    for (int i = 0; i < 16; i++) {
      af[i]  = (_Float16)As[m_loc][kk[i]];
      bf0[i] = (_Float16)Bs[kk[i]][m_loc];
      bf1[i] = (_Float16)Bs[kk[i]][16 + m_loc];
    }
    acc0 = wmma_f16(af, bf0, acc0);
    acc1 = wmma_f16(af, bf1, acc1);
    __syncthreads();
  }
#pragma unroll
  for (int r = 0; r < 8; r++) {
    int grow = row0 + r + ((lane & 16) ? 8 : 0);
    if (grow < M) {
      int g0 = col0 + m_loc, g1 = col0 + 16 + m_loc;
      if (g0 < N) {
        float v = acc0[r];
        if (bias) v += bias[g0];
        Cb[(long)grow * ldc + g0] = actf(v, act);
      }
      if (g1 < N) {
        float v = acc1[r];
        if (bias) v += bias[g1];
        Cb[(long)grow * ldc + g1] = actf(v, act);
      }
    }
  }
}

// ---------------------------------------------------------------------------
// Embedding gather: out[i,:] = emb[idx[i],:]
// ---------------------------------------------------------------------------
__global__ void k_gather(const int* __restrict__ idx, const float* __restrict__ emb,
                         float* __restrict__ out, int n_tok, int ed) {
  int i = blockIdx.x;
  if (i >= n_tok) return;
  int id = idx[i];
  const float* src = emb + (long)id * ed;
  float* dst = out + (long)i * ed;
  for (int e = threadIdx.x; e < ed; e += blockDim.x) dst[e] = src[e];
}

// ---------------------------------------------------------------------------
// GRU scan (one direction). One workgroup, state kept in LDS, per-step
// gh = h @ Whh + bhh computed with WMMA tiles distributed over 8 waves.
// xp = x@Wih + bih is precomputed (Mb, L, 3H). Output ys[(m*L+t)*ld+off+j].
// ---------------------------------------------------------------------------
__global__ void k_gru_scan(const float* __restrict__ xp, const float* __restrict__ Whh,
                           const float* __restrict__ bhh, float* __restrict__ ys,
                           int Mb, int L, int H, int ld_out, int col_off, int rev) {
  extern __shared__ float smem[];
  int H3 = 3 * H;
  int ntiles = (H3 + 15) / 16;
  int mtiles = (Mb + 15) / 16;
  int Mp = mtiles * 16;
  int N3p = ntiles * 16;
  float* hS = smem;                    // Mp * H
  float* gS = smem + (long)Mp * H;     // Mp * N3p
  int tid = threadIdx.x, nthr = blockDim.x;
  int lane = tid & 31, wid = tid >> 5, nwaves = nthr >> 5;
  int kk[16]; frag_k_list(lane, kk);
  int m_loc = lane & 15;
  for (int i = tid; i < Mp * H; i += nthr) hS[i] = 0.f;
  __syncthreads();
  int kchunks = (H + 31) / 32;
  for (int s = 0; s < L; s++) {
    int t = rev ? (L - 1 - s) : s;
    for (int tile = wid; tile < ntiles * mtiles; tile += nwaves) {
      int tn = tile % ntiles, tm = tile / ntiles;
      int col0 = tn * 16, row0 = tm * 16;
      int gn = col0 + m_loc;
      bool nok = (gn < H3);
      v8f acc = {};
#pragma unroll 2
      for (int kc = 0; kc < kchunks; kc++) {
        int kb = kc * 32;
        v16h af, bf;
        if (kb + 32 <= H && nok) {      // interior: no guards
#pragma unroll
          for (int i = 0; i < 16; i++) {
            int k = kb + kk[i];
            af[i] = (_Float16)hS[(long)(row0 + m_loc) * H + k];
            bf[i] = (_Float16)Whh[(long)k * H3 + gn];
          }
        } else {
#pragma unroll
          for (int i = 0; i < 16; i++) {
            int k = kb + kk[i];
            float av = (k < H) ? hS[(long)(row0 + m_loc) * H + k] : 0.f;
            float bv = (k < H && nok) ? Whh[(long)k * H3 + gn] : 0.f;
            af[i] = (_Float16)av; bf[i] = (_Float16)bv;
          }
        }
        acc = wmma_f16(af, bf, acc);
      }
#pragma unroll
      for (int r = 0; r < 8; r++) {
        int grow = row0 + r + ((lane & 16) ? 8 : 0);
        gS[(long)grow * N3p + gn] = acc[r] + (nok ? bhh[gn] : 0.f);
      }
    }
    __syncthreads();
    for (int idx = tid; idx < Mb * H; idx += nthr) {
      int m = idx / H, j = idx - m * H;
      const float* xt = xp + ((long)m * L + t) * H3;
      float ir = xt[j], iz = xt[H + j], in = xt[2 * H + j];
      float hr = gS[(long)m * N3p + j];
      float hz = gS[(long)m * N3p + H + j];
      float hn = gS[(long)m * N3p + 2 * H + j];
      float r = 1.f / (1.f + __expf(-(ir + hr)));
      float z = 1.f / (1.f + __expf(-(iz + hz)));
      float n = tanhf(in + r * hn);
      float h2 = (1.f - z) * n + z * hS[(long)m * H + j];
      hS[(long)m * H + j] = h2;
      ys[((long)m * L + t) * ld_out + col_off + j] = h2;
    }
    __syncthreads();
  }
}

// ---------------------------------------------------------------------------
// Additive attention scores (sc): out[b,q,p] = sum_e tanh(U[b,p,e]+V[b,q,e])*vv[e]
// ---------------------------------------------------------------------------
__global__ void k_add_score(const float* __restrict__ U, const float* __restrict__ V,
                            const float* __restrict__ vv, float* __restrict__ out,
                            int Q, int P, int E) {
  int q = blockIdx.x, b = blockIdx.y;
  extern __shared__ float shv[];   // 2*E
  float* Vr = shv; float* vcs = shv + E;
  for (int e = threadIdx.x; e < E; e += blockDim.x) {
    Vr[e] = V[((long)b * Q + q) * E + e];
    vcs[e] = vv[e];
  }
  __syncthreads();
  for (int p = threadIdx.x; p < P; p += blockDim.x) {
    const float* Ur = U + ((long)b * P + p) * E;
    float sum = 0.f;
    for (int e = 0; e < E; e++) sum += tanhf(Ur[e] + Vr[e]) * vcs[e];
    out[((long)b * Q + q) * P + p] = sum;
  }
}

// ---------------------------------------------------------------------------
// Pairwise score (sd/sm/ss): out[b,q,p] = sum_e tanh((op(Hq_p,Hp_q) @ W)[e])*v[e]
// op: 0 = elementwise mul, 1 = subtract. One wave per 16 p-rows, WMMA over
// E n-tiles x D/32 k-steps, fused tanh*v + shfl reduction.
// ---------------------------------------------------------------------------
__global__ void k_pair_score(const float* __restrict__ Hq, const float* __restrict__ Hp,
                             const float* __restrict__ W, const float* __restrict__ v,
                             float* __restrict__ out,
                             int Q, int P, int D, int E, int op) {
  int pt = blockIdx.x, q = blockIdx.y, b = blockIdx.z;
  __shared__ float hpq[1024];
  __shared__ float As[16][33];
  int lane = threadIdx.x;
  for (int e = lane; e < D; e += 32) hpq[e] = Hp[((long)b * Q + q) * D + e];
  __syncthreads();
  int kk[16]; frag_k_list(lane, kk);
  int m_loc = lane & 15;
  int p0 = pt * 16;
  float rowsum[8];
#pragma unroll
  for (int r = 0; r < 8; r++) rowsum[r] = 0.f;
  for (int nt = 0; nt < E / 16; nt++) {
    v8f acc = {};
#pragma unroll 2
    for (int kc = 0; kc < D; kc += 32) {
      for (int idx = lane; idx < 16 * 32; idx += 32) {
        int r = idx >> 5, c = idx & 31;
        int gp = p0 + r, gk = kc + c;
        float hv = (gp < P) ? Hq[((long)b * P + gp) * D + gk] : 0.f;
        float pv = hpq[gk];
        As[r][c] = op ? (hv - pv) : (hv * pv);
      }
      __syncthreads();
      v16h af, bf;
#pragma unroll
      for (int i = 0; i < 16; i++) {
        af[i] = (_Float16)As[m_loc][kk[i]];
        bf[i] = (_Float16)W[(long)(kc + kk[i]) * E + nt * 16 + m_loc];
      }
      acc = wmma_f16(af, bf, acc);
      __syncthreads();
    }
    float vv = v[nt * 16 + m_loc];
#pragma unroll
    for (int r = 0; r < 8; r++) {
      float val = tanhf(acc[r]) * vv;
      for (int off = 8; off >= 1; off >>= 1) val += __shfl_xor(val, off, 16);
      rowsum[r] += val;
    }
  }
  if (m_loc == 0) {
    int base = (lane & 16) ? 8 : 0;
    for (int r = 0; r < 8; r++) {
      int gp = p0 + base + r;
      if (gp < P) out[((long)b * Q + q) * P + gp] = rowsum[r];
    }
  }
}

// ---------------------------------------------------------------------------
// Row softmax in place, R rows of C columns. blockDim must be power of 2.
// ---------------------------------------------------------------------------
__global__ void k_softmax(float* __restrict__ x, int R, int C) {
  int r = blockIdx.x;
  extern __shared__ float red[];
  int tid = threadIdx.x;
  float* row = x + (long)r * C;
  float mx = -1e30f;
  for (int c = tid; c < C; c += blockDim.x) mx = fmaxf(mx, row[c]);
  red[tid] = mx; __syncthreads();
  for (int s = blockDim.x / 2; s > 0; s >>= 1) {
    if (tid < s) red[tid] = fmaxf(red[tid], red[tid + s]);
    __syncthreads();
  }
  mx = red[0]; __syncthreads();
  float sum = 0.f;
  for (int c = tid; c < C; c += blockDim.x) { float e = __expf(row[c] - mx); row[c] = e; sum += e; }
  red[tid] = sum; __syncthreads();
  for (int s = blockDim.x / 2; s > 0; s >>= 1) {
    if (tid < s) red[tid] += red[tid + s];
    __syncthreads();
  }
  float inv = 1.f / red[0];
  for (int c = tid; c < C; c += blockDim.x) row[c] *= inv;
}

// out[r*ostride+ooff] = sum_e A[r,e]*v[e]
__global__ void k_rowdot(const float* __restrict__ A, const float* __restrict__ v,
                         float* __restrict__ out, int E, int ostride, int ooff) {
  int r = blockIdx.x, tid = threadIdx.x;
  __shared__ float red[256];
  float s = 0.f;
  for (int e = tid; e < E; e += blockDim.x) s += A[(long)r * E + e] * v[e];
  red[tid] = s; __syncthreads();
  for (int k = blockDim.x / 2; k > 0; k >>= 1) {
    if (tid < k) red[tid] += red[tid + k];
    __syncthreads();
  }
  if (tid == 0) out[(long)r * ostride + ooff] = red[0];
}

// C[r,:] = concat(A[r,:D1], B[r,:D2])
__global__ void k_concat2(const float* __restrict__ A, const float* __restrict__ B,
                          float* __restrict__ C, int D1, int D2) {
  int r = blockIdx.x;
  int D = D1 + D2;
  for (int d = threadIdx.x; d < D; d += blockDim.x)
    C[(long)r * D + d] = (d < D1) ? A[(long)r * D1 + d] : B[(long)r * D2 + (d - D1)];
}

__global__ void k_mul(const float* __restrict__ a, const float* __restrict__ b,
                      float* __restrict__ c, long n) {
  long i = (long)blockIdx.x * blockDim.x + threadIdx.x;
  if (i < n) c[i] = a[i] * b[i];
}

// out[r,d] = sum_t w[r,t] * X[(r*T+t)*D + d]
__global__ void k_wsum(const float* __restrict__ w, const float* __restrict__ X,
                       float* __restrict__ out, int T, int D) {
  int r = blockIdx.x;
  for (int d = threadIdx.x; d < D; d += blockDim.x) {
    float s = 0.f;
    for (int t = 0; t < T; t++) s += w[(long)r * T + t] * X[((long)r * T + t) * D + d];
    out[(long)r * D + d] = s;
  }
}

// rep[r,d] = sum_k w4[r*4+k] * h_k[r,d]
__global__ void k_agg_combine(const float* __restrict__ w4,
                              const float* __restrict__ h0, const float* __restrict__ h1,
                              const float* __restrict__ h2, const float* __restrict__ h3,
                              float* __restrict__ rep, int D) {
  int r = blockIdx.x;
  for (int d = threadIdx.x; d < D; d += blockDim.x) {
    long o = (long)r * D + d;
    rep[o] = w4[r * 4 + 0] * h0[o] + w4[r * 4 + 1] * h1[o] +
             w4[r * 4 + 2] * h2[o] + w4[r * 4 + 3] * h3[o];
  }
}

// out[b,q] = sum_e (U1[b,q,e]+V2[b,e])*vp[e]   (no tanh, per reference)
__global__ void k_sp_score(const float* __restrict__ U1, const float* __restrict__ V2,
                           const float* __restrict__ vp, float* __restrict__ out,
                           int Q, int E) {
  int q = blockIdx.x, b = blockIdx.y, tid = threadIdx.x;
  __shared__ float red[256];
  float s = 0.f;
  for (int e = tid; e < E; e += blockDim.x)
    s += (U1[((long)b * Q + q) * E + e] + V2[(long)b * E + e]) * vp[e];
  red[tid] = s; __syncthreads();
  for (int k = blockDim.x / 2; k > 0; k >>= 1) {
    if (tid < k) red[tid] += red[tid + k];
    __syncthreads();
  }
  if (tid == 0) out[(long)b * Q + q] = red[0];
}

// logits[b,k] = a_emb[b,k,:]·enc[b,:]; loss = -mean_b log_softmax(logits)[b,0]
__global__ void k_loss(const float* __restrict__ a_emb, const float* __restrict__ enc,
                       float* __restrict__ out, int Bn, int ED) {
  __shared__ float red[256];
  __shared__ float logits[32];
  int tid = threadIdx.x;
  for (int bk = 0; bk < Bn * 3; bk++) {
    int b = bk / 3;
    float s = 0.f;
    for (int d = tid; d < ED; d += blockDim.x)
      s += a_emb[(long)bk * ED + d] * enc[(long)b * ED + d];
    red[tid] = s; __syncthreads();
    for (int k = blockDim.x / 2; k > 0; k >>= 1) {
      if (tid < k) red[tid] += red[tid + k];
      __syncthreads();
    }
    if (tid == 0) logits[bk] = red[0];
    __syncthreads();
  }
  if (tid == 0) {
    float loss = 0.f;
    for (int b = 0; b < Bn; b++) {
      float l0 = logits[b * 3], l1 = logits[b * 3 + 1], l2 = logits[b * 3 + 2];
      float m = fmaxf(l0, fmaxf(l1, l2));
      float se = __expf(l0 - m) + __expf(l1 - m) + __expf(l2 - m);
      loss += (l0 - m) - __logf(se);
    }
    out[0] = -loss / (float)Bn;
  }
}

// ===========================================================================
// Host orchestration
// ===========================================================================
static inline int ceil16(int x) { return (x + 15) / 16; }
static inline int ceil32(int x) { return (x + 31) / 32; }

extern "C" void kernel_launch(void* const* d_in, const int* in_sizes, int n_in,
                              void* d_out, int out_size, void* d_ws, size_t ws_size,
                              hipStream_t stream) {
  if (n_in < 92) return;   // expected flattened param count

  const int B = 8, LQ = 32, LP = 256, LA = 24, EMB = 300, E = 256;
  const int D2 = 2 * E, D4 = 4 * E, H3 = 3 * E;   // 512, 1024, 768
  const int Ha = EMB / 2;                          // 150

  // Input layout: 0=query 1=passage 2=answer 3=is_train, then params in
  // dict-insertion order (GRU dicts: f{Wih,Whh,bih,bhh} then b{...}).
  auto P = [&](int i) { return (const float*)d_in[i]; };
  const int* query   = (const int*)d_in[0];
  const int* passage = (const int*)d_in[1];
  const int* answer  = (const int*)d_in[2];
  const float* emb = P(4);
  // param indices
  const int A_ENC = 5, Q_ENC = 13, P_ENC = 21, A_ATT = 29;
  const int WC1 = 30, WC2 = 31, VC = 32, WB = 33, WD = 34, VD = 35;
  const int WM = 36, VM = 37, WS = 38, VS = 39;
  const int GRU_AGG = 40, WGC = 48, GRU_HTC = 49, WGB = 57, GRU_HTB = 58;
  const int WGD = 66, GRU_HTD = 67, WGM = 75, GRU_HTM = 76;
  const int W_AGG = 84, V_AGG = 85, WQ = 86, VQ = 87;
  const int WP1 = 88, WP2 = 89, VP = 90, WPRED = 91;

  // bump allocator over workspace
  float* ws = (float*)d_ws;
  size_t woff = 0;
  auto alloc = [&](size_t n) { float* p = ws + woff; woff += (n + 63) & ~(size_t)63; return p; };

  auto gemm = [&](const float* A, const float* Bm, const float* bias, float* C,
                  int M, int N, int K, int lda, int ldb, int ldc,
                  long sA, long sB, long sC, int batch, int transB, int act) {
    dim3 g(ceil32(N), ceil16(M), batch);
    k_gemm<<<g, 32, 0, stream>>>(A, Bm, bias, C, M, N, K, lda, ldb, ldc,
                                 sA, sB, sC, transB, act);
  };

  auto bigru = [&](const float* x, int pb, int Mb, int L, int din, int H,
                   float* ys, float* xpf, float* xpb) {
    int h3 = 3 * H;
    gemm(x, P(pb + 0), P(pb + 2), xpf, Mb * L, h3, din, din, h3, h3, 0, 0, 0, 1, 0, 0);
    gemm(x, P(pb + 4), P(pb + 6), xpb, Mb * L, h3, din, din, h3, h3, 0, 0, 0, 1, 0, 0);
    int mt = ceil16(Mb), nt = ceil16(h3);
    size_t shm = ((size_t)mt * 16 * H + (size_t)mt * 16 * nt * 16) * sizeof(float);
    k_gru_scan<<<1, 256, shm, stream>>>(xpf, P(pb + 1), P(pb + 3), ys, Mb, L, H, 2 * H, 0, 0);
    k_gru_scan<<<1, 256, shm, stream>>>(xpb, P(pb + 5), P(pb + 7), ys, Mb, L, H, 2 * H, H, 1);
  };

  // ---- embeddings -------------------------------------------------------
  float* qe = alloc((size_t)B * LQ * EMB);
  float* pe = alloc((size_t)B * LP * EMB);
  float* ae = alloc((size_t)B * 3 * LA * EMB);
  k_gather<<<B * LQ, 128, 0, stream>>>(query, emb, qe, B * LQ, EMB);
  k_gather<<<B * LP, 128, 0, stream>>>(passage, emb, pe, B * LP, EMB);
  k_gather<<<B * 3 * LA, 128, 0, stream>>>(answer, emb, ae, B * 3 * LA, EMB);

  // ---- answer encoder + attention --------------------------------------
  float* xaf = alloc((size_t)24 * LA * 3 * Ha);
  float* xab = alloc((size_t)24 * LA * 3 * Ha);
  float* a_h = alloc((size_t)24 * LA * EMB);
  bigru(ae, A_ENC, 24, LA, EMB, Ha, a_h, xaf, xab);
  float* a_sc = alloc((size_t)24 * LA);
  k_rowdot<<<24 * LA, 256, 0, stream>>>(a_h, P(A_ATT), a_sc, EMB, 1, 0);
  k_softmax<<<24, 32, 32 * sizeof(float), stream>>>(a_sc, 24, LA);
  float* a_emb = alloc((size_t)24 * EMB);
  k_wsum<<<24, 256, 0, stream>>>(a_sc, a_h, a_emb, LA, EMB);

  // ---- passage / query encoders ----------------------------------------
  float* xqf = alloc((size_t)B * LP * H3);
  float* xqb = alloc((size_t)B * LP * H3);
  float* hq  = alloc((size_t)B * LP * D2);
  bigru(pe, Q_ENC, B, LP, EMB, E, hq, xqf, xqb);
  float* xhf = alloc((size_t)B * LQ * H3);
  float* xhb = alloc((size_t)B * LQ * H3);
  float* hp  = alloc((size_t)B * LQ * D2);
  bigru(qe, P_ENC, B, LQ, EMB, E, hp, xhf, xhb);

  // ---- sc (concat/additive attention) ----------------------------------
  float* U  = alloc((size_t)B * LP * E);
  float* Vq = alloc((size_t)B * LQ * E);
  gemm(hq, P(WC1), nullptr, U,  B * LP, E, D2, D2, E, E, 0, 0, 0, 1, 0, 0);
  gemm(hp, P(WC2), nullptr, Vq, B * LQ, E, D2, D2, E, E, 0, 0, 0, 1, 0, 0);
  float* sc = alloc((size_t)B * LQ * LP);
  k_add_score<<<dim3(LQ, B), 256, 2 * E * sizeof(float), stream>>>(U, Vq, P(VC), sc, LQ, LP, E);
  k_softmax<<<B * LQ, 256, 256 * sizeof(float), stream>>>(sc, B * LQ, LP);
  float* qtc = alloc((size_t)B * LQ * D2);
  gemm(sc, hq, nullptr, qtc, LQ, D2, LP, LP, D2, D2,
       (long)LQ * LP, (long)LP * D2, (long)LQ * D2, B, 0, 0);

  // ---- sb (bilinear) ----------------------------------------------------
  float* hqWb = alloc((size_t)B * LP * D2);
  gemm(hq, P(WB), nullptr, hqWb, B * LP, D2, D2, D2, D2, D2, 0, 0, 0, 1, 0, 0);
  float* sb = alloc((size_t)B * LQ * LP);
  gemm(hp, hqWb, nullptr, sb, LQ, LP, D2, D2, D2, LP,
       (long)LQ * D2, (long)LP * D2, (long)LQ * LP, B, 1, 0);
  k_softmax<<<B * LQ, 256, 256 * sizeof(float), stream>>>(sb, B * LQ, LP);
  float* qtb = alloc((size_t)B * LQ * D2);
  gemm(sb, hq, nullptr, qtb, LQ, D2, LP, LP, D2, D2,
       (long)LQ * LP, (long)LP * D2, (long)LQ * D2, B, 0, 0);

  // ---- sd (dot), sm (minus), ss (self) ---------------------------------
  float* sd = alloc((size_t)B * LQ * LP);
  k_pair_score<<<dim3(LP / 16, LQ, B), 32, 0, stream>>>(hq, hp, P(WD), P(VD), sd, LQ, LP, D2, E, 0);
  k_softmax<<<B * LQ, 256, 256 * sizeof(float), stream>>>(sd, B * LQ, LP);
  float* qtd = alloc((size_t)B * LQ * D2);
  gemm(sd, hq, nullptr, qtd, LQ, D2, LP, LP, D2, D2,
       (long)LQ * LP, (long)LP * D2, (long)LQ * D2, B, 0, 0);

  float* smx = alloc((size_t)B * LQ * LP);
  k_pair_score<<<dim3(LP / 16, LQ, B), 32, 0, stream>>>(hq, hp, P(WM), P(VM), smx, LQ, LP, D2, E, 1);
  k_softmax<<<B * LQ, 256, 256 * sizeof(float), stream>>>(smx, B * LQ, LP);
  float* qtm = alloc((size_t)B * LQ * D2);
  gemm(smx, hq, nullptr, qtm, LQ, D2, LP, LP, D2, D2,
       (long)LQ * LP, (long)LP * D2, (long)LQ * D2, B, 0, 0);

  float* ssb = alloc((size_t)B * LQ * LQ);
  k_pair_score<<<dim3(LQ / 16, LQ, B), 32, 0, stream>>>(hp, hp, P(WS), P(VS), ssb, LQ, LQ, D2, E, 0);
  k_softmax<<<B * LQ, 32, 32 * sizeof(float), stream>>>(ssb, B * LQ, LQ);
  float* qts = alloc((size_t)B * LQ * D2);
  gemm(ssb, hp, nullptr, qts, LQ, D2, LQ, LQ, D2, D2,
       (long)LQ * LQ, (long)LQ * D2, (long)LQ * D2, B, 0, 0);
  (void)qts;  // computed per reference; not consumed downstream

  // ---- gated GRUs -------------------------------------------------------
  float* xcat = alloc((size_t)B * LQ * D4);
  float* gbuf = alloc((size_t)B * LQ * D4);
  float* xgf  = alloc((size_t)B * LQ * H3);
  float* xgb  = alloc((size_t)B * LQ * H3);
  float* ht[4];
  const float* qts_in[4] = { qtc, qtb, qtd, qtm };
  const int wg_idx[4]  = { WGC, WGB, WGD, WGM };
  const int gru_idx[4] = { GRU_HTC, GRU_HTB, GRU_HTD, GRU_HTM };
  for (int k = 0; k < 4; k++) {
    ht[k] = alloc((size_t)B * LQ * D2);
    k_concat2<<<B * LQ, 256, 0, stream>>>(qts_in[k], hp, xcat, D2, D2);
    gemm(xcat, P(wg_idx[k]), nullptr, gbuf, B * LQ, D4, D4, D4, D4, D4, 0, 0, 0, 1, 0, 1);
    long n = (long)B * LQ * D4;
    k_mul<<<(int)((n + 255) / 256), 256, 0, stream>>>(gbuf, xcat, gbuf, n);
    bigru(gbuf, gru_idx[k], B, LQ, D4, E, ht[k], xgf, xgb);
  }

  // ---- aggregation attention + gru_agg ----------------------------------
  float* T   = alloc((size_t)B * LQ * E);
  float* sa4 = alloc((size_t)B * LQ * 4);
  for (int k = 0; k < 4; k++) {
    gemm(ht[k], P(W_AGG), nullptr, T, B * LQ, E, D2, D2, E, E, 0, 0, 0, 1, 0, 2);
    k_rowdot<<<B * LQ, 256, 0, stream>>>(T, P(V_AGG), sa4, E, 4, k);
  }
  k_softmax<<<B * LQ, 32, 32 * sizeof(float), stream>>>(sa4, B * LQ, 4);
  float* rep = alloc((size_t)B * LQ * D2);
  k_agg_combine<<<B * LQ, 256, 0, stream>>>(sa4, ht[0], ht[1], ht[2], ht[3], rep, D2);
  float* agg_rep = alloc((size_t)B * LQ * D2);
  bigru(rep, GRU_AGG, B, LQ, D2, E, agg_rep, xgf, xgb);

  // ---- question summary rq ----------------------------------------------
  float* Tq  = alloc((size_t)B * LP * E);
  float* sqv = alloc((size_t)B * LP);
  gemm(hq, P(WQ), nullptr, Tq, B * LP, E, D2, D2, E, E, 0, 0, 0, 1, 0, 2);
  k_rowdot<<<B * LP, 256, 0, stream>>>(Tq, P(VQ), sqv, E, 1, 0);
  k_softmax<<<B, 256, 256 * sizeof(float), stream>>>(sqv, B, LP);
  float* rq = alloc((size_t)B * D2);
  k_wsum<<<B, 256, 0, stream>>>(sqv, hq, rq, LP, D2);

  // ---- pointer + prediction ---------------------------------------------
  float* U1  = alloc((size_t)B * LQ * E);
  float* V2  = alloc((size_t)B * E);
  gemm(agg_rep, P(WP1), nullptr, U1, B * LQ, E, D2, D2, E, E, 0, 0, 0, 1, 0, 0);
  gemm(rq, P(WP2), nullptr, V2, B, E, D2, D2, E, E, 0, 0, 0, 1, 0, 0);
  float* spv = alloc((size_t)B * LQ);
  k_sp_score<<<dim3(LQ, B), 256, 0, stream>>>(U1, V2, P(VP), spv, LQ, E);
  k_softmax<<<B, 32, 32 * sizeof(float), stream>>>(spv, B, LQ);
  float* rp = alloc((size_t)B * D2);
  k_wsum<<<B, 256, 0, stream>>>(spv, agg_rep, rp, LQ, D2);
  float* encb = alloc((size_t)B * EMB);
  gemm(rp, P(WPRED), nullptr, encb, B, EMB, D2, D2, EMB, EMB, 0, 0, 0, 1, 0, 3);

  // ---- loss --------------------------------------------------------------
  k_loss<<<1, 256, 0, stream>>>(a_emb, encb, (float*)d_out, B, EMB);

  (void)in_sizes; (void)out_size; (void)ws_size;
}
